// Tensorf_11725260718372
// MI455X (gfx1250) — matrix operand order repaired and checked by reading, not compiled, not run on hardware
//
#include <hip/hip_runtime.h>
#include <hip/hip_bf16.h>

typedef _Float16 f16;
typedef _Float16 v16h __attribute__((ext_vector_type(16)));
typedef _Float16 v8h  __attribute__((ext_vector_type(8)));
typedef _Float16 v4h  __attribute__((ext_vector_type(4)));
typedef _Float16 h2t  __attribute__((ext_vector_type(2)));
typedef float    v8f  __attribute__((ext_vector_type(8)));

#define NVOXC 128
#define RSIG  48
#define RFEAT 144
#define PDIM  27

// workspace layout (units: halfs). All region bases 16B-aligned in bytes.
#define BT_K     160                      // 144 ranks padded to 160 (5 K-chunks of 32)
#define BT_OFF   0                        // BT[32][160]    (P=27 padded to 32 cols)
#define W1T_OFF  (BT_OFF + 32*BT_K)       // W1T[128][128]  (K=120 row carries b1)
#define W2T_OFF  (W1T_OFF + 128*128)      // W2T[128][160]  (K=128 row carries b2)
#define W3T_OFF  (W2T_OFF + 128*160)      // W3T[16][160]   (K=128 row carries b3)
#define SGT_OFF  (W3T_OFF + 16*160)       // sigmaT[3][128][64]  rank-major f16
#define FTT_OFF  (SGT_OFF + 3*128*64)     // featT [3][128][160] rank-major f16
#define WS_HALFS (FTT_OFF + 3*128*160)

union V16 { v16h v; v8h h2[2]; };
union V8u { v8h  v; v4h q[2]; };
union V4u { v4h  v; h2t d[2]; };

__device__ __forceinline__ v8h splat8(f16 x) {
  v8h r = {x, x, x, x, x, x, x, x};
  return r;
}

__device__ __forceinline__ v8f wmma16(v16h a, v16h b, v8f c) {
  // D = A(16x32 f16) * B(32x16 f16) + C(16x16 f32)
  return __builtin_amdgcn_wmma_f32_16x16x32_f16(false, a, false, b, (short)0, c,
                                                false, false);
}

__device__ __forceinline__ v16h load_afrag(const f16* row, int kc, int hh) {
  V16 u;
  u.h2[0] = *(const v8h*)(row + 32 * kc + 16 * hh);
  u.h2[1] = *(const v8h*)(row + 32 * kc + 16 * hh + 8);
  return u.v;
}

// ------- prep: transpose + f16-convert weights (bias folded into K) + CP tables -------
__global__ void prep_tables(const float* __restrict__ B,  const float* __restrict__ W1,
                            const float* __restrict__ W2, const float* __restrict__ W3,
                            const float* __restrict__ b1, const float* __restrict__ b2,
                            const float* __restrict__ b3,
                            const float* __restrict__ sigma, const float* __restrict__ feat,
                            f16* __restrict__ ws) {
  int t = blockIdx.x * blockDim.x + threadIdx.x;
  if (t < 32 * BT_K) {                       // BT[n][k] = B[k][n],  B is (144,27)
    int n = t / BT_K, k = t % BT_K;
    ws[BT_OFF + t] = (f16)((n < PDIM && k < RFEAT) ? B[k * PDIM + n] : 0.0f);
    return;
  }
  t -= 32 * BT_K;
  if (t < 128 * 128) {                       // W1T[n][k]; k=120 row = b1[n]
    int n = t / 128, k = t % 128;
    float v = (k < 120) ? W1[k * 128 + n] : ((k == 120) ? b1[n] : 0.0f);
    ws[W1T_OFF + t] = (f16)v;
    return;
  }
  t -= 128 * 128;
  if (t < 128 * 160) {                       // W2T[n][k]; k=128 row = b2[n]
    int n = t / 160, k = t % 160;
    float v = (k < 128) ? W2[k * 128 + n] : ((k == 128) ? b2[n] : 0.0f);
    ws[W2T_OFF + t] = (f16)v;
    return;
  }
  t -= 128 * 160;
  if (t < 16 * 160) {                        // W3T[n][k]; k=128 row = b3[n]
    int n = t / 160, k = t % 160;
    float v = 0.0f;
    if (n < 3) v = (k < 128) ? W3[k * 3 + n] : ((k == 128) ? b3[n] : 0.0f);
    ws[W3T_OFF + t] = (f16)v;
    return;
  }
  t -= 16 * 160;
  if (t < 3 * 128 * 64) {                    // sigmaT[a][v][r] = sigma[a][r][v]
    int a = t / (128 * 64), rem = t % (128 * 64);
    int v = rem / 64, r = rem % 64;
    ws[SGT_OFF + t] = (f16)((r < RSIG) ? sigma[(a * RSIG + r) * NVOXC + v] : 0.0f);
    return;
  }
  t -= 3 * 128 * 64;
  if (t < 3 * 128 * 160) {                   // featT[a][v][r] = feat[a][r][v]
    int a = t / (128 * 160), rem = t % (128 * 160);
    int v = rem / 160, r = rem % 160;
    ws[FTT_OFF + t] = (f16)((r < RFEAT) ? feat[(a * RFEAT + r) * NVOXC + v] : 0.0f);
  }
}

// ---------------- fused TensoRF kernel: 1 wave = 32 points (2 M-tiles) ----------------
__launch_bounds__(128)
__global__ void tensorf_fused(const float* __restrict__ xyz,  const float* __restrict__ dirs,
                              const float* __restrict__ voxel,const f16* __restrict__ ws,
                              float* __restrict__ out, int Np) {
  __shared__ f16 hbuf[4][2][16][160];        // per wave: two 16x160 activation tiles
                                             // (rows K=128..159: ones row for bias-in-K)
  const int lane = threadIdx.x & 31;
  const int wave = threadIdx.x >> 5;
  const int m    = lane & 15;                // row (point) within tile
  const int hh   = lane >> 4;                // half: 0 -> K 0..15 of each 32-chunk, 1 -> K 16..31
  const int base = (blockIdx.x * 4 + wave) * 32;

  const f16* sgT = ws + SGT_OFF;             // [3][128][64]
  const f16* ftT = ws + FTT_OFF;             // [3][128][160]

  // ---- voxel snap: paired-column base + adjusted weight (edges fold to 0/1) ----
  int bidx[2][3];
  f16 wqh[2][3];
#pragma unroll
  for (int u = 0; u < 2; ++u) {
    int p = base + u * 16 + m;
    p = (p < Np) ? p : (Np - 1);
#pragma unroll
    for (int a = 0; a < 3; ++a) {
      float x = xyz[p * 3 + a];
      int idx = (int)ceilf(x * 127.0f);
      idx = idx < 0 ? 0 : (idx > 128 ? 128 : idx);
      int il = idx - 1; il = il < 0 ? 0 : (il > 127 ? 127 : il);
      int ir = idx > 127 ? 127 : idx;
      float vl = voxel[a * NVOXC + il], vr = voxel[a * NVOXC + ir];
      float w = (x - vl) / (vr - vl + 1e-6f);
      float wadj = (il == ir) ? ((il == 0) ? 0.0f : 1.0f) : w;
      wqh[u][a]  = (f16)wadj;
      bidx[u][a] = (il < 126) ? il : 126;
    }
  }

  // ---- density: packed-f16 lerp/product, 8 ranks per b128; pk tree-reduce ----
#pragma unroll
  for (int u = 0; u < 2; ++u) {
    v8h acc8 = {};
#pragma unroll
    for (int i = 0; i < 3; ++i) {
      const int oct = 2 * i + hh;            // ranks 8*oct .. 8*oct+7 (< 48)
      v8h pr;
#pragma unroll
      for (int a = 0; a < 3; ++a) {
        const f16* col = sgT + (a * NVOXC + bidx[u][a]) * 64 + 8 * oct;
        v8h lo = *(const v8h*)col;
        v8h hi = *(const v8h*)(col + 64);
        v8h val = lo + (hi - lo) * splat8(wqh[u][a]);
        pr = (a == 0) ? val : pr * val;
      }
      acc8 = acc8 + pr;
    }
    V8u t8; t8.v = acc8;
    V4u t4; t4.v = t8.q[0] + t8.q[1];
    h2t t2 = t4.d[0] + t4.d[1];
    float s = (float)t2[0] + (float)t2[1];
    s += __shfl_xor(s, 16);
    int pG = base + u * 16 + m;
    if (hh == 0 && pG < Np) {
      float v = s - 5.0f;                    // SIGMA_BIAS
      out[pG] = (v > 20.0f) ? v : log1pf(__expf(v));
    }
  }

  // ---- feature rank-products: packed f16 straight into WMMA A fragments ----
  v16h afr[2][5];
#pragma unroll
  for (int u = 0; u < 2; ++u) {
#pragma unroll
    for (int kc = 0; kc < 5; ++kc) {
      V16 av;
#pragma unroll
      for (int seg = 0; seg < 2; ++seg) {
        const int oct = 4 * kc + 2 * hh + seg;   // K octet index
        v8h pr = {};
        if (oct < RFEAT / 8) {                   // octets 18,19 are zero padding
          v8h p;
#pragma unroll
          for (int a = 0; a < 3; ++a) {
            const f16* col = ftT + (a * NVOXC + bidx[u][a]) * BT_K + 8 * oct;
            v8h lo = *(const v8h*)col;
            v8h hi = *(const v8h*)(col + BT_K);
            v8h val = lo + (hi - lo) * splat8(wqh[u][a]);
            p = (a == 0) ? val : p * val;
          }
          pr = p;
        }
        av.h2[seg] = pr;
      }
      afr[u][kc] = av.v;
    }
  }

  // ---- feats = fprod(16x144) @ B(144x27): shared B fragment, fused encode ----
#pragma unroll
  for (int t = 0; t < 2; ++t) {
    v8f acc[2] = {{}, {}};
    const int n = t * 16 + m;
#pragma unroll
    for (int kc = 0; kc < 5; ++kc) {
      v16h bfr = *(const v16h*)(ws + BT_OFF + n * BT_K + 32 * kc + 16 * hh);
      acc[0] = wmma16(afr[0][kc], bfr, acc[0]);
      acc[1] = wmma16(afr[1][kc], bfr, acc[1]);
    }
    if (n < PDIM) {                          // encode: sin/cos at K = n,27+n,54+n,81+n
#pragma unroll
      for (int u = 0; u < 2; ++u) {
#pragma unroll
        for (int i = 0; i < 8; ++i) {
          int M = i + 8 * hh;
          float f = acc[u][i];
          hbuf[wave][u][M][n]      = (f16)__sinf(f);
          hbuf[wave][u][M][27 + n] = (f16)__cosf(f);
          hbuf[wave][u][M][54 + n] = (f16)__sinf(2.0f * f);
          hbuf[wave][u][M][81 + n] = (f16)__cosf(2.0f * f);
        }
      }
    }
  }
  // dirs fill K=108..119; K=120 = 1.0 (b1 row); K=121..127 = 0; K=128..159 ones-row/zeros
  {
    int p = base + hh * 16 + m;
    p = (p < Np) ? p : (Np - 1);
    float d[3] = { dirs[p * 3 + 0], dirs[p * 3 + 1], dirs[p * 3 + 2] };
#pragma unroll
    for (int j = 0; j < 3; ++j) {
      hbuf[wave][hh][m][108 + j] = (f16)__sinf(d[j]);
      hbuf[wave][hh][m][111 + j] = (f16)__cosf(d[j]);
      hbuf[wave][hh][m][114 + j] = (f16)__sinf(2.0f * d[j]);
      hbuf[wave][hh][m][117 + j] = (f16)__cosf(2.0f * d[j]);
    }
    hbuf[wave][hh][m][120] = (f16)1.0f;      // L1 bias row
#pragma unroll
    for (int j = 121; j < 128; ++j) hbuf[wave][hh][m][j] = (f16)0.0f;
    v8h ones = {};
    ones[0] = (f16)1.0f;                     // K=128 -> L2/L3 bias row, rest zero
    *(v8h*)&hbuf[wave][hh][m][128] = ones;
    v8h z = {};
    *(v8h*)&hbuf[wave][hh][m][136] = z;
    *(v8h*)&hbuf[wave][hh][m][144] = z;
    *(v8h*)&hbuf[wave][hh][m][152] = z;
  }
  __syncthreads();

  // ---- layer 1: h(16x128) @ W1(128x128), bias in K=120, leaky = max(v, 0.01v) ----
  {
    v16h a[2][4];
#pragma unroll
    for (int u = 0; u < 2; ++u)
#pragma unroll
      for (int kc = 0; kc < 4; ++kc) a[u][kc] = load_afrag(&hbuf[wave][u][m][0], kc, hh);
    __syncthreads();
#pragma unroll
    for (int t = 0; t < 8; ++t) {
      v8f acc[2] = {{}, {}};
      const int n = t * 16 + m;
#pragma unroll
      for (int kc = 0; kc < 4; ++kc) {
        v16h bfr = *(const v16h*)(ws + W1T_OFF + n * 128 + 32 * kc + 16 * hh);
        acc[0] = wmma16(a[0][kc], bfr, acc[0]);
        acc[1] = wmma16(a[1][kc], bfr, acc[1]);
      }
#pragma unroll
      for (int u = 0; u < 2; ++u)
#pragma unroll
        for (int i = 0; i < 8; ++i) {
          float v = acc[u][i];
          hbuf[wave][u][i + 8 * hh][n] = (f16)fmaxf(v, 0.01f * v);
        }
    }
    __syncthreads();
  }

  // ---- layer 2: h(16x160 incl ones row) @ W2T(128x160), bias in K=128 ----
  {
    v16h a[2][5];
#pragma unroll
    for (int u = 0; u < 2; ++u)
#pragma unroll
      for (int kc = 0; kc < 5; ++kc) a[u][kc] = load_afrag(&hbuf[wave][u][m][0], kc, hh);
    __syncthreads();
#pragma unroll
    for (int t = 0; t < 8; ++t) {
      v8f acc[2] = {{}, {}};
      const int n = t * 16 + m;
#pragma unroll
      for (int kc = 0; kc < 5; ++kc) {
        v16h bfr = *(const v16h*)(ws + W2T_OFF + n * 160 + 32 * kc + 16 * hh);
        acc[0] = wmma16(a[0][kc], bfr, acc[0]);
        acc[1] = wmma16(a[1][kc], bfr, acc[1]);
      }
#pragma unroll
      for (int u = 0; u < 2; ++u)
#pragma unroll
        for (int i = 0; i < 8; ++i) {
          float v = acc[u][i];
          hbuf[wave][u][i + 8 * hh][n] = (f16)fmaxf(v, 0.01f * v);
        }
    }
    __syncthreads();
  }

  // ---- layer 3: @ W3T(16x160), bias in K=128, sigmoid, scatter rgb ----
  {
    v16h a[2][5];
#pragma unroll
    for (int u = 0; u < 2; ++u)
#pragma unroll
      for (int kc = 0; kc < 5; ++kc) a[u][kc] = load_afrag(&hbuf[wave][u][m][0], kc, hh);
    v8f acc[2] = {{}, {}};
#pragma unroll
    for (int kc = 0; kc < 5; ++kc) {
      v16h bfr = *(const v16h*)(ws + W3T_OFF + m * 160 + 32 * kc + 16 * hh);
      acc[0] = wmma16(a[0][kc], bfr, acc[0]);
      acc[1] = wmma16(a[1][kc], bfr, acc[1]);
    }
    if (m < 3) {
#pragma unroll
      for (int u = 0; u < 2; ++u)
#pragma unroll
        for (int i = 0; i < 8; ++i) {
          int p = base + u * 16 + i + 8 * hh;
          if (p < Np) {
            out[Np + p * 3 + m] = 1.0f / (1.0f + __expf(-acc[u][i]));
          }
        }
    }
  }
}

extern "C" void kernel_launch(void* const* d_in, const int* in_sizes, int n_in,
                              void* d_out, int out_size, void* d_ws, size_t ws_size,
                              hipStream_t stream) {
  const float* xyz   = (const float*)d_in[0];
  const float* dirs  = (const float*)d_in[1];
  const float* voxel = (const float*)d_in[2];
  const float* sigma = (const float*)d_in[3];
  const float* feat  = (const float*)d_in[4];
  const float* B     = (const float*)d_in[5];
  const float* W1    = (const float*)d_in[6];
  const float* b1    = (const float*)d_in[7];
  const float* W2    = (const float*)d_in[8];
  const float* b2    = (const float*)d_in[9];
  const float* W3    = (const float*)d_in[10];
  const float* b3    = (const float*)d_in[11];
  float* out = (float*)d_out;
  f16* ws = (f16*)d_ws;

  int Np = in_sizes[0] / 3;

  prep_tables<<<(WS_HALFS + 255) / 256, 256, 0, stream>>>(B, W1, W2, W3, b1, b2, b3,
                                                          sigma, feat, ws);

  int nblocks = (Np + 127) / 128;            // 1 block = 4 waves = 128 points
  tensorf_fused<<<nblocks, 128, 0, stream>>>(xyz, dirs, voxel, ws, out, Np);
}
